// LSDQN_12867722019089
// MI455X (gfx1250) — matrix operand order repaired and checked by reading, not compiled, not run on hardware
//
#include <hip/hip_runtime.h>
#include <hip/hip_bf16.h>

typedef _Float16 half_t;
typedef __attribute__((ext_vector_type(16))) _Float16 v16h;
typedef __attribute__((ext_vector_type(4)))  _Float16 v4h;
typedef __attribute__((ext_vector_type(8)))  float    v8f;
typedef __attribute__((ext_vector_type(4)))  unsigned int v4u;
typedef __attribute__((ext_vector_type(8)))  int v8i;
typedef __attribute__((ext_vector_type(4)))  int v4i;

#define NN 100000
#define EE 800000
#define HH 128
#define NG 64
#define HOPS 3

static __device__ __forceinline__ void atomAddF(float* p, float v) {
    unsafeAtomicAdd(p, v);   // native global_atomic_add_f32
}

// ---------------- zero ----------------
__global__ void k_zero(uint32_t* p, long n) {
    long i = (long)blockIdx.x * blockDim.x + threadIdx.x;
    if (i < n) p[i] = 0u;
}

// ---------------- pack l1_w (3 hops) into WMMA B-fragment layout (f16) ----
// idx = ((hop*4 + t)*8 + c)*512 + l*16 + j
// k = t*32 + (l>=16 ? 8 : 0) + (j & 7) + (j>=8 ? 16 : 0) ; n = c*16 + (l & 15)
__global__ void k_pack_l1(const float* __restrict__ l1w, half_t* __restrict__ pk) {
    int idx = blockIdx.x * blockDim.x + threadIdx.x;
    if (idx >= HOPS * 4 * 8 * 32 * 16) return;
    int j = idx & 15;
    int l = (idx >> 4) & 31;
    int c = (idx >> 9) & 7;
    int t = (idx >> 12) & 3;
    int hop = idx >> 14;
    int k = t * 32 + ((l & 16) ? 8 : 0) + (j & 7) + ((j & 8) ? 16 : 0);
    int n = c * 16 + (l & 15);
    pk[idx] = (half_t)l1w[hop * HH * HH + k * HH + n];
}

// ---------------- degree ----------------
__global__ void k_deg(const int* __restrict__ row, float* __restrict__ degf) {
    int e = blockIdx.x * blockDim.x + threadIdx.x;
    if (e < EE) atomAddF(degf + row[e], 1.0f);
}

// ---------------- fold sp/ac into vectors, bias scalar ----------------
__global__ void k_fold(const float* __restrict__ sp_w, const float* __restrict__ sp_b,
                       const float* __restrict__ ac_w, const float* __restrict__ ac_b,
                       const float* __restrict__ last_w, const float* __restrict__ last_b,
                       float* __restrict__ vsp, float* __restrict__ vac, float* __restrict__ c0) {
    int t = threadIdx.x;
    if (t < HH) {
        float s = 0.f;
        for (int j = 0; j < HH; ++j) s += sp_w[t * HH + j] * last_w[j];
        vsp[t] = s;
    }
    {
        float s = 0.f;
        for (int j = 0; j < HH; ++j) s += ac_w[t * HH + j] * last_w[HH + j];
        vac[t] = s;
    }
    if (t == 0) {
        float s = last_b[0];
        for (int j = 0; j < HH; ++j) s += sp_b[j] * last_w[j] + ac_b[j] * last_w[HH + j];
        c0[0] = s;
    }
}

// ---------------- edge gather + scatter-add (one wave per edge) ----------
__global__ void k_edge(const int* __restrict__ row, const int* __restrict__ col,
                       const float* __restrict__ ea, const half_t* __restrict__ u16,
                       float* __restrict__ uagg, float* __restrict__ s3,
                       const float* __restrict__ l3w, const float* __restrict__ l3b, int hop) {
    int g = blockIdx.x * blockDim.x + threadIdx.x;
    int e = g >> 5, lane = g & 31;
    if (e >= EE) return;
    int r = row[e], c = col[e];
    float w = ea[e];
    v4h uh = *(const v4h*)(u16 + (size_t)c * HH + lane * 4);
    float* dst = uagg + (size_t)r * HH + lane * 4;
    atomAddF(dst + 0, w * (float)uh[0]);
    atomAddF(dst + 1, w * (float)uh[1]);
    atomAddF(dst + 2, w * (float)uh[2]);
    atomAddF(dst + 3, w * (float)uh[3]);
    if (lane == 0) {
        float eev = fmaxf(w * l3w[hop] + l3b[hop], 0.f);
        atomAddF(s3 + r, eev);
    }
}

// ---- issue one TDM 1-D tile load: nbytes (multiple of 8) global -> LDS ----
static __device__ __forceinline__ void tdm_load_to_lds(const void* gptr, uint32_t lds_off,
                                                       uint32_t nbytes) {
    uint64_t ga = (uint64_t)(uintptr_t)gptr;
    uint32_t units = nbytes >> 3;                       // data_size = 8B
    v4u g0;
    g0[0] = 1u;                                         // count=1, user mode
    g0[1] = lds_off;                                    // lds_addr
    g0[2] = (uint32_t)ga;                               // global_addr[31:0]
    g0[3] = ((uint32_t)(ga >> 32) & 0x01FFFFFFu) | (2u << 30);  // addr[56:32] | type=2
    v8i g1;
    g1[0] = (int)(3u << 16);                            // data_size=8B, no pad/multicast
    g1[1] = (int)((units & 0xFFFFu) << 16);             // tensor_dim0[15:0]
    g1[2] = (int)(((units >> 16) & 0xFFFFu) | (1u << 16)); // tensor_dim0[31:16], tensor_dim1=1
    g1[3] = (int)((units & 0xFFFFu) << 16);             // tile_dim0
    g1[4] = 0;                                          // tile_dim1/2 unused
    g1[5] = (int)units;                                 // tensor_dim0_stride[31:0]
    g1[6] = 0;
    g1[7] = 0;
    v4i z4 = {0, 0, 0, 0};
#if defined(__clang_major__) && (__clang_major__ >= 23)
    v8i z8 = {0, 0, 0, 0, 0, 0, 0, 0};
    __builtin_amdgcn_tensor_load_to_lds(g0, g1, z4, z4, z8, 0);
#else
    __builtin_amdgcn_tensor_load_to_lds(g0, g1, z4, z4, 0);
#endif
}

// ---------------- node update GEMM (WMMA f16->f32) + fused epilogue ------
// one wave: 16 node rows x 128 out cols; block = 8 waves = 128 rows
// weights (32KB packed B fragments) are DMA'd into LDS once per block (TDM)
__global__ void __launch_bounds__(256)
k_node(const float* __restrict__ uagg, const float* __restrict__ s3,
       const float* __restrict__ degf, const float* __restrict__ x,
       const half_t* __restrict__ l1pk,
       const float* __restrict__ l0w, const float* __restrict__ l0b,
       const float* __restrict__ l1b, const float* __restrict__ l2w,
       const float* __restrict__ l2b, half_t* __restrict__ u16) {
    __shared__ half_t sB[HH * HH];          // 32 KB packed B fragments
    __shared__ float xs[128], m3s[128], ids[128];
    int tid = threadIdx.x;
    int rblk = blockIdx.x * 128;

    // async weight DMA from wave 0, overlapped with the scalar prologue
    if (tid < 32) {
        tdm_load_to_lds(l1pk, (uint32_t)(uintptr_t)&sB[0], HH * HH * 2);
    }
    if (tid < 128) {
        int r = rblk + tid;
        if (r < NN) {
            float d = degf[r];
            float id = d > 0.f ? 1.f / d : 0.f;
            xs[tid] = x[r];
            ids[tid] = id;
            m3s[tid] = s3[r] * id;
        } else {
            xs[tid] = 0.f; ids[tid] = 0.f; m3s[tid] = 0.f;
        }
    }
    if (tid < 32) {
        __builtin_amdgcn_s_wait_tensorcnt(0);
    }
    __syncthreads();

    int wave = tid >> 5, lane = tid & 31;
    int lr0 = wave * 16 + (lane & 15);
    int r = rblk + lr0;
    int sr = r < NN ? r : 0;
    float scale = ids[lr0];
    int koff = (lane < 16) ? 0 : 8;
    const float* arow = uagg + (size_t)sr * HH;

    v8f acc[8] = {};
    #pragma unroll
    for (int t = 0; t < 4; ++t) {
        int kb = t * 32 + koff;
        float4 f0 = *(const float4*)(arow + kb);
        float4 f1 = *(const float4*)(arow + kb + 4);
        float4 f2 = *(const float4*)(arow + kb + 16);
        float4 f3 = *(const float4*)(arow + kb + 20);
        v16h a;
        a[0]  = (half_t)(f0.x * scale); a[1]  = (half_t)(f0.y * scale);
        a[2]  = (half_t)(f0.z * scale); a[3]  = (half_t)(f0.w * scale);
        a[4]  = (half_t)(f1.x * scale); a[5]  = (half_t)(f1.y * scale);
        a[6]  = (half_t)(f1.z * scale); a[7]  = (half_t)(f1.w * scale);
        a[8]  = (half_t)(f2.x * scale); a[9]  = (half_t)(f2.y * scale);
        a[10] = (half_t)(f2.z * scale); a[11] = (half_t)(f2.w * scale);
        a[12] = (half_t)(f3.x * scale); a[13] = (half_t)(f3.y * scale);
        a[14] = (half_t)(f3.z * scale); a[15] = (half_t)(f3.w * scale);
        #pragma unroll
        for (int c = 0; c < 8; ++c) {
            v16h b = *(const v16h*)(&sB[(((t * 8 + c) * 32) + lane) * 16]);
            acc[c] = __builtin_amdgcn_wmma_f32_16x16x32_f16(
                false, a, false, b, (short)0, acc[c], false, false);
        }
    }

    #pragma unroll
    for (int c = 0; c < 8; ++c) {
        int h = c * 16 + (lane & 15);
        float w0h = l0w[h], w2h = l2w[h];
        float bias_c = l0b[h] + l1b[h] + l2b[h];
        #pragma unroll
        for (int v = 0; v < 8; ++v) {
            int lr = wave * 16 + v + ((lane < 16) ? 0 : 8);
            int rr = rblk + lr;
            if (rr < NN) {
                float val = acc[c][v] + xs[lr] * w0h + m3s[lr] * w2h + bias_c;
                u16[(size_t)rr * HH + h] = (half_t)fmaxf(val, 0.f);
            }
        }
    }
}

// ---------------- per-graph pooling (one wave per node) ----------------
__global__ void k_pool(const half_t* __restrict__ u16, const float* __restrict__ x,
                       const int* __restrict__ batch,
                       float* __restrict__ pool0, float* __restrict__ pool1,
                       float* __restrict__ cnt0, float* __restrict__ cnt1) {
    int g = blockIdx.x * blockDim.x + threadIdx.x;
    int n = g >> 5, lane = g & 31;
    if (n >= NN) return;
    int b = batch[n];
    float xb = x[n];
    v4h uh = *(const v4h*)(u16 + (size_t)n * HH + lane * 4);
    float* base = (xb > 0.5f ? pool1 : pool0) + b * HH + lane * 4;
    atomAddF(base + 0, (float)uh[0]);
    atomAddF(base + 1, (float)uh[1]);
    atomAddF(base + 2, (float)uh[2]);
    atomAddF(base + 3, (float)uh[3]);
    if (lane == 0) atomAddF((xb > 0.5f ? cnt1 : cnt0) + b, 1.0f);
}

__global__ void k_hc(const float* __restrict__ pool0, const float* __restrict__ pool1,
                     const float* __restrict__ cnt0, const float* __restrict__ cnt1,
                     float* __restrict__ hc0, float* __restrict__ hc1) {
    int i = blockIdx.x * blockDim.x + threadIdx.x;
    if (i >= NG * HH) return;
    int b = i >> 7;
    hc0[i] = cnt0[b] > 0.f ? pool0[i] / cnt0[b] : 0.f;
    hc1[i] = cnt1[b] > 0.f ? pool1[i] / cnt1[b] : 0.f;
}

// ---------------- per-graph mat-vec + scalar precompute ----------------
__global__ void k_graphvec(const float* __restrict__ att_w, const float* __restrict__ att_b,
                           const float* __restrict__ hc0, const float* __restrict__ hc1,
                           const float* __restrict__ vsp, const float* __restrict__ vac,
                           float* __restrict__ G0, float* __restrict__ G1,
                           float* __restrict__ gscal) {
    __shared__ float h0[128], h1[128], g0s[256], g1s[256];
    int b = blockIdx.x, t = threadIdx.x;
    if (t < 128) { h0[t] = hc0[b * HH + t]; h1[t] = hc1[b * HH + t]; }
    __syncthreads();
    {
        float s0 = 0.f, s1 = 0.f;
        for (int j = 0; j < HH; ++j) {
            float w = att_w[t * HH + j];
            s0 += w * h0[j]; s1 += w * h1[j];
        }
        g0s[t] = s0; g1s[t] = s1;
        G0[b * 256 + t] = s0; G1[b * 256 + t] = s1;
    }
    __syncthreads();
    if (t < 10) {
        float s = 0.f;
        for (int j = 0; j < HH; ++j) {
            switch (t) {
                case 0: s += att_b[j] * h0[j]; break;
                case 1: s += att_b[j] * h1[j]; break;
                case 2: s += g0s[HH + j] * h0[j]; break;
                case 3: s += g0s[HH + j] * h1[j]; break;
                case 4: s += g1s[HH + j] * h0[j]; break;
                case 5: s += g1s[HH + j] * h1[j]; break;
                case 6: s += vsp[j] * h0[j]; break;
                case 7: s += vsp[j] * h1[j]; break;
                case 8: s += vac[HH + j] * h0[j]; break;
                case 9: s += vac[HH + j] * h1[j]; break;
            }
        }
        gscal[b * 16 + t] = s;
    }
}

// ---------------- readout (one wave per node, 3 dots + softmax) ---------
__global__ void k_readout(const half_t* __restrict__ u16, const float* __restrict__ x,
                          const int* __restrict__ batch,
                          const float* __restrict__ G0, const float* __restrict__ G1,
                          const float* __restrict__ vac, const float* __restrict__ gscal,
                          const float* __restrict__ c0, float* __restrict__ out) {
    int g = blockIdx.x * blockDim.x + threadIdx.x;
    int n = g >> 5, lane = g & 31;
    if (n >= NN) return;
    int b = batch[n];
    float xb = x[n];
    const float* G0b = G0 + b * 256;
    const float* G1b = G1 + b * 256;
    v4h uh = *(const v4h*)(u16 + (size_t)n * HH + lane * 4);
    float d0 = 0.f, d1 = 0.f, da = 0.f;
    #pragma unroll
    for (int j = 0; j < 4; ++j) {
        int h = lane * 4 + j;
        float uv = (float)uh[j];
        d0 += uv * G0b[h];
        d1 += uv * G1b[h];
        da += uv * vac[h];
    }
    #pragma unroll
    for (int o = 16; o > 0; o >>= 1) {
        d0 += __shfl_xor(d0, o, 32);
        d1 += __shfl_xor(d1, o, 32);
        da += __shfl_xor(da, o, 32);
    }
    if (lane == 0) {
        const float* gs = gscal + b * 16;
        float w0 = d0 + (xb > 0.5f ? gs[2] : gs[3]) + gs[0];
        float w1 = d1 + (xb > 0.5f ? gs[4] : gs[5]) + gs[1];
        float m = fmaxf(w0, w1);
        float z0 = __expf(w0 - m), z1 = __expf(w1 - m);
        float s0 = z0 / (z0 + z1), s1 = 1.f - s0;
        out[n] = s0 * gs[6] + s1 * gs[7] + da + (xb > 0.5f ? gs[8] : gs[9]) + c0[0];
    }
}

extern "C" void kernel_launch(void* const* d_in, const int* in_sizes, int n_in,
                              void* d_out, int out_size, void* d_ws, size_t ws_size,
                              hipStream_t stream) {
    const float* x      = (const float*)d_in[0];
    const int*   ei     = (const int*)d_in[1];
    const int*   row    = ei;
    const int*   col    = ei + EE;
    const float* ea     = (const float*)d_in[2];
    const int*   batch  = (const int*)d_in[3];
    const float* l0w    = (const float*)d_in[5];
    const float* l0b    = (const float*)d_in[6];
    const float* l1w    = (const float*)d_in[7];
    const float* l1b    = (const float*)d_in[8];
    const float* l2w    = (const float*)d_in[9];
    const float* l2b    = (const float*)d_in[10];
    const float* l3w    = (const float*)d_in[11];
    const float* l3b    = (const float*)d_in[12];
    const float* att_w  = (const float*)d_in[13];
    const float* att_b  = (const float*)d_in[14];
    const float* sp_w   = (const float*)d_in[15];
    const float* sp_b   = (const float*)d_in[16];
    const float* ac_w   = (const float*)d_in[17];
    const float* ac_b   = (const float*)d_in[18];
    const float* last_w = (const float*)d_in[19];
    const float* last_b = (const float*)d_in[20];
    float* q = (float*)d_out;

    char* ws = (char*)d_ws;
    size_t off = 0;
    auto carve = [&](size_t bytes) -> char* {
        char* p = ws + off;
        off = (off + bytes + 255) & ~(size_t)255;
        return p;
    };
    half_t* u16  = (half_t*)carve((size_t)NN * HH * 2);
    float*  uagg = (float*)carve((size_t)NN * HH * 4);
    float*  s3   = (float*)carve((size_t)NN * 4);
    float*  degf = (float*)carve((size_t)NN * 4);
    half_t* l1pk = (half_t*)carve((size_t)HOPS * HH * HH * 2);
    float*  vsp  = (float*)carve(HH * 4);
    float*  vac  = (float*)carve(2 * HH * 4);
    float*  c0   = (float*)carve(4);
    float*  poolblk = (float*)carve((size_t)(2 * NG * HH + 2 * NG) * 4);
    float*  pool0 = poolblk;
    float*  pool1 = poolblk + NG * HH;
    float*  cnt0  = poolblk + 2 * NG * HH;
    float*  cnt1  = poolblk + 2 * NG * HH + NG;
    float*  hc0  = (float*)carve((size_t)NG * HH * 4);
    float*  hc1  = (float*)carve((size_t)NG * HH * 4);
    float*  G0   = (float*)carve((size_t)NG * 256 * 4);
    float*  G1   = (float*)carve((size_t)NG * 256 * 4);
    float*  gscal= (float*)carve((size_t)NG * 16 * 4);

    const int TB = 256;
    auto blocks = [](long n) { return (int)((n + 255) / 256); };

    // init
    k_zero<<<blocks((long)NN * HH / 2), TB, 0, stream>>>((uint32_t*)u16, (long)NN * HH / 2);
    k_zero<<<blocks(NN), TB, 0, stream>>>((uint32_t*)degf, NN);
    k_pack_l1<<<blocks(HOPS * 16384), TB, 0, stream>>>(l1w, l1pk);
    k_deg<<<blocks(EE), TB, 0, stream>>>(row, degf);
    k_fold<<<1, TB, 0, stream>>>(sp_w, sp_b, ac_w, ac_b, last_w, last_b, vsp, vac, c0);

    // hop loop
    for (int hop = 0; hop < HOPS; ++hop) {
        k_zero<<<blocks((long)NN * HH), TB, 0, stream>>>((uint32_t*)uagg, (long)NN * HH);
        k_zero<<<blocks(NN), TB, 0, stream>>>((uint32_t*)s3, NN);
        k_edge<<<(int)((long)EE * 32 / 256), TB, 0, stream>>>(
            row, col, ea, u16, uagg, s3, l3w, l3b, hop);
        k_node<<<(NN + 127) / 128, TB, 0, stream>>>(
            uagg, s3, degf, x, l1pk + (size_t)hop * HH * HH,
            l0w + hop * HH, l0b + hop * HH, l1b + hop * HH,
            l2w + hop * HH, l2b + hop * HH, u16);
    }

    // pooling + readout
    k_zero<<<blocks(2 * NG * HH + 2 * NG), TB, 0, stream>>>(
        (uint32_t*)poolblk, 2 * NG * HH + 2 * NG);
    k_pool<<<(int)((long)NN * 32 / 256), TB, 0, stream>>>(
        u16, x, batch, pool0, pool1, cnt0, cnt1);
    k_hc<<<blocks(NG * HH), TB, 0, stream>>>(pool0, pool1, cnt0, cnt1, hc0, hc1);
    k_graphvec<<<NG, TB, 0, stream>>>(att_w, att_b, hc0, hc1, vsp, vac, G0, G1, gscal);
    k_readout<<<(int)((long)NN * 32 / 256), TB, 0, stream>>>(
        u16, x, batch, G0, G1, vac, gscal, c0, q);
    (void)in_sizes; (void)n_in; (void)out_size; (void)ws_size;
}
